// SparseInvariantMultiQueryAttention_3685081940022
// MI455X (gfx1250) — matrix operand order repaired and checked by reading, not compiled, not run on hardware
//
#include <hip/hip_runtime.h>
#include <hip/hip_bf16.h>
#include <stdint.h>

// Problem constants (from reference)
#define NNODE 8192
#define KNB   32
#define CLC   384
#define CPC   128
#define HH    4
#define SS    32
#define PPT   8
#define NPROJ 336   // 128 (q) + 32 (k) + 32 (v) + 96 (qp) + 24 (kp) + 24 (vp)
#define NCAT  736   // 128 (local_up) + 512 (pair_up) + 96 (pt)
#define NBLK  3     // N-tiles per wave strip (16x48); 48 | 336 and 48 | 384

typedef __attribute__((ext_vector_type(16))) __bf16 v16bf;
typedef __attribute__((ext_vector_type(8)))  float  v8f;

union Frag { v16bf v; uint32_t u[8]; };

// ---------------------------------------------------------------------------
// WMMA bf16 GEMM:  C[M x N] = A[M x K] * B[K x N] (+bias per column)
// A row-major bf16; BT is B transposed (N x K row-major) so each WMMA
// B-fragment 32-bit load grabs a contiguous (k, k+1) bf16 pair (vectorizes
// to global_load_b128). Each wave computes a 16x48 strip: the A fragment is
// reused across 3 accumulators -> 3 independent v_wmma per K-step.
// Grid sized so there are no partial tiles (EXEC stays all-ones for WMMA).
// ---------------------------------------------------------------------------
__global__ __launch_bounds__(128)
void wmma_gemm_bf16nt(const __bf16* __restrict__ A,
                      const __bf16* __restrict__ BT,
                      const float* __restrict__ bias,
                      float* __restrict__ C,
                      int M, int Nc, int Kd)
{
  const int wave = threadIdx.x >> 5;
  const int lane = threadIdx.x & 31;
  const int stripsN = Nc / (16 * NBLK);
  const int strip = blockIdx.x * (blockDim.x >> 5) + wave;
  const int tm  = strip / stripsN;
  const int tn0 = (strip - tm * stripsN) * NBLK;
  if (tm * 16 >= M) return;                 // wave-uniform; never taken (exact grid)

  const int hi = lane >> 4;                 // half-wave select
  const int mr = lane & 15;                 // A row within tile == B/C column within tile
  const __bf16* arow  = A  + (size_t)(tm * 16 + mr) * Kd;
  const __bf16* brow0 = BT + (size_t)((tn0 + 0) * 16 + mr) * Kd;
  const __bf16* brow1 = BT + (size_t)((tn0 + 1) * 16 + mr) * Kd;
  const __bf16* brow2 = BT + (size_t)((tn0 + 2) * 16 + mr) * Kd;
  // A 16x32 layout (matches B): lanes 0-15 K=0..15, lanes 16-31 K=16..31; v_i -> K = kb0+2i, +1
  const int ka0 = hi * 8;
  const int ka1 = 16 + hi * 8;
  // B 32x16 layout: lanes 0-15 K=0..15, lanes 16-31 K=16..31; v_i -> K = kb0+2i, +1
  const int kb0 = hi * 16;

  v8f acc0 = {0.f, 0.f, 0.f, 0.f, 0.f, 0.f, 0.f, 0.f};
  v8f acc1 = acc0, acc2 = acc0;
  for (int k0 = 0; k0 < Kd; k0 += 32) {
    Frag af, b0, b1, b2;
#pragma unroll
    for (int i = 0; i < 8; ++i) {
      af.u[i]     = *(const uint32_t*)(arow + k0 + kb0 + 2 * i);
    }
#pragma unroll
    for (int i = 0; i < 8; ++i) {
      b0.u[i] = *(const uint32_t*)(brow0 + k0 + kb0 + 2 * i);
      b1.u[i] = *(const uint32_t*)(brow1 + k0 + kb0 + 2 * i);
      b2.u[i] = *(const uint32_t*)(brow2 + k0 + kb0 + 2 * i);
    }
    acc0 = __builtin_amdgcn_wmma_f32_16x16x32_bf16(false, af.v, false, b0.v,
                                                   (short)0, acc0, false, false);
    acc1 = __builtin_amdgcn_wmma_f32_16x16x32_bf16(false, af.v, false, b1.v,
                                                   (short)0, acc1, false, false);
    acc2 = __builtin_amdgcn_wmma_f32_16x16x32_bf16(false, af.v, false, b2.v,
                                                   (short)0, acc2, false, false);
  }

  const float bd0 = bias ? bias[(tn0 + 0) * 16 + mr] : 0.0f;
  const float bd1 = bias ? bias[(tn0 + 1) * 16 + mr] : 0.0f;
  const float bd2 = bias ? bias[(tn0 + 2) * 16 + mr] : 0.0f;
#pragma unroll
  for (int i = 0; i < 8; ++i) {
    const int row = tm * 16 + i + hi * 8;   // ISA 16x16 f32 C/D layout
    float* crow = C + (size_t)row * Nc;
    crow[(tn0 + 0) * 16 + mr] = acc0[i] + bd0;
    crow[(tn0 + 1) * 16 + mr] = acc1[i] + bd1;
    crow[(tn0 + 2) * 16 + mr] = acc2[i] + bd2;
  }
}

// ---------------------------------------------------------------------------
// Prep kernels
// ---------------------------------------------------------------------------
__global__ void k_f32_to_bf16(const float* __restrict__ in, __bf16* __restrict__ out, int n)
{
  int i = blockIdx.x * blockDim.x + threadIdx.x;
  if (i < n) out[i] = (__bf16)in[i];
}

__global__ void k_build_wcat(const float* __restrict__ Wq,  const float* __restrict__ Wk,
                             const float* __restrict__ Wv,  const float* __restrict__ Wqp,
                             const float* __restrict__ Wkp, const float* __restrict__ Wvp,
                             const float* __restrict__ bq,  const float* __restrict__ bk,
                             const float* __restrict__ bv,  const float* __restrict__ bqp,
                             const float* __restrict__ bkp, const float* __restrict__ bvp,
                             __bf16* __restrict__ WcatT, float* __restrict__ biascat)
{
  int idx = blockIdx.x * blockDim.x + threadIdx.x;
  if (idx < NPROJ) {
    int n = idx; float b;
    if      (n < 128) b = bq[n];
    else if (n < 160) b = bk[n - 128];
    else if (n < 192) b = bv[n - 160];
    else if (n < 288) b = bqp[n - 192];
    else if (n < 312) b = bkp[n - 288];
    else              b = bvp[n - 312];
    biascat[n] = b;
  }
  if (idx >= NPROJ * CLC) return;
  int n = idx / CLC;
  int k = idx - n * CLC;
  float v;
  if      (n < 128) v = Wq [(size_t)k * 128 + n];
  else if (n < 160) v = Wk [(size_t)k * 32  + (n - 128)];
  else if (n < 192) v = Wv [(size_t)k * 32  + (n - 160)];
  else if (n < 288) v = Wqp[(size_t)k * 96  + (n - 192)];
  else if (n < 312) v = Wkp[(size_t)k * 24  + (n - 288)];
  else              v = Wvp[(size_t)k * 24  + (n - 312)];
  WcatT[(size_t)n * CLC + k] = (__bf16)v;
}

__global__ void k_build_woutT(const float* __restrict__ Wout, __bf16* __restrict__ WoutT)
{
  int idx = blockIdx.x * blockDim.x + threadIdx.x;
  if (idx >= CLC * NCAT) return;
  int n = idx / NCAT, k = idx - n * NCAT;
  WoutT[idx] = (__bf16)Wout[(size_t)k * CLC + n];
}

// ---------------------------------------------------------------------------
// Projection epilogue: per-node LN (q per head, k) + rigid-frame application.
// One thread per node (tiny, bandwidth-bound).
// ---------------------------------------------------------------------------
__global__ void k_epilogue(const float* __restrict__ proj, const float* __restrict__ frames,
                           float* __restrict__ q_ln, float* __restrict__ k_ln,
                           float* __restrict__ qp_g, float* __restrict__ kp_g,
                           float* __restrict__ vp_g)
{
  int n = blockIdx.x * blockDim.x + threadIdx.x;
  if (n >= NNODE) return;
  const float* row = proj + (size_t)n * NPROJ;

  float rot[9], tr[3];
#pragma unroll
  for (int a = 0; a < 3; ++a) {
#pragma unroll
    for (int b = 0; b < 3; ++b) rot[a * 3 + b] = frames[(size_t)n * 16 + a * 4 + b];
    tr[a] = frames[(size_t)n * 16 + a * 4 + 3];
  }

  for (int h = 0; h < HH; ++h) {          // q LN per head
    const float* x = row + h * SS;
    float m = 0.f;
    for (int s = 0; s < SS; ++s) m += x[s];
    m *= (1.0f / SS);
    float v = 0.f;
    for (int s = 0; s < SS; ++s) { float d = x[s] - m; v += d * d; }
    float rs = rsqrtf(v * (1.0f / SS) + 1e-5f);
    for (int s = 0; s < SS; ++s)
      q_ln[(size_t)n * 128 + h * SS + s] = (x[s] - m) * rs;
  }
  {                                        // k LN
    const float* x = row + 128;
    float m = 0.f;
    for (int s = 0; s < SS; ++s) m += x[s];
    m *= (1.0f / SS);
    float v = 0.f;
    for (int s = 0; s < SS; ++s) { float d = x[s] - m; v += d * d; }
    float rs = rsqrtf(v * (1.0f / SS) + 1e-5f);
    for (int s = 0; s < SS; ++s)
      k_ln[(size_t)n * SS + s] = (x[s] - m) * rs;
  }
  for (int h = 0; h < HH; ++h)             // qp = rot*p + trans
    for (int p = 0; p < PPT; ++p) {
      const float* r = row + 192 + h * 24 + p * 3;
      for (int a = 0; a < 3; ++a)
        qp_g[(size_t)n * 96 + h * 24 + p * 3 + a] =
            rot[a*3+0]*r[0] + rot[a*3+1]*r[1] + rot[a*3+2]*r[2] + tr[a];
    }
  for (int p = 0; p < PPT; ++p) {          // kp, vp
    const float* r1 = row + 288 + p * 3;
    const float* r2 = row + 312 + p * 3;
    for (int a = 0; a < 3; ++a) {
      kp_g[(size_t)n * 24 + p * 3 + a] =
          rot[a*3+0]*r1[0] + rot[a*3+1]*r1[1] + rot[a*3+2]*r1[2] + tr[a];
      vp_g[(size_t)n * 24 + p * 3 + a] =
          rot[a*3+0]*r2[0] + rot[a*3+1]*r2[1] + rot[a*3+2]*r2[2] + tr[a];
    }
  }
}

// ---------------------------------------------------------------------------
// Fused attention: one 128-thread block per node.
// `pair[n]` (32x128, 16KB) staged once in LDS and reused for both the bias
// (pair@Wb) and pair_up (attn.pair) -> single HBM pass over the 134MB tensor.
// Outputs written directly as bf16 into the 736-wide concat row.
// ---------------------------------------------------------------------------
#define PLD 132   // padded LDS stride (128 % 64 == 0 would serialize banks)

__global__ __launch_bounds__(128)
void k_attention(const float* __restrict__ pair, const float* __restrict__ frames,
                 const int* __restrict__ neighbours, const unsigned char* __restrict__ mask,
                 const float* __restrict__ proj, const float* __restrict__ q_ln,
                 const float* __restrict__ k_ln, const float* __restrict__ qp_g,
                 const float* __restrict__ kp_g, const float* __restrict__ vp_g,
                 const float* __restrict__ Wb, const float* __restrict__ gamma,
                 __bf16* __restrict__ cat)
{
  __shared__ float s_pair[KNB * PLD];
  __shared__ float s_q[128];
  __shared__ float s_logits[KNB * HH];
  __shared__ float s_attn[KNB * HH];
  __shared__ float s_Wb[CPC * HH];
  __shared__ int   s_idx[KNB];
  __shared__ float s_rot[9], s_tr[3];
  __shared__ unsigned char s_mask[KNB];
  __shared__ float s_pt[HH * PPT * 3];

  const int n = blockIdx.x;
  const int t = threadIdx.x;

  if (t < KNB) {
    s_idx[t]  = neighbours[(size_t)n * KNB + t];
    s_mask[t] = mask[(size_t)n * KNB + t];
  }
  s_q[t] = q_ln[(size_t)n * 128 + t];
  if (t < 9) s_rot[t] = frames[(size_t)n * 16 + (t / 3) * 4 + (t % 3)];
  if (t < 3) s_tr[t]  = frames[(size_t)n * 16 + t * 4 + 3];
  for (int i = t; i < CPC * HH; i += 128) s_Wb[i] = Wb[i];
  for (int k = 0; k < KNB; ++k)
    s_pair[k * PLD + t] = pair[((size_t)n * KNB + k) * CPC + t];
  __syncthreads();

  // logits: thread -> (k, h)
  {
    const int k = t >> 2;
    const int h = t & 3;
    const int nb = s_idx[k];
    const float* qr = s_q + h * SS;
    const float* kr = k_ln + (size_t)nb * SS;
    float qk = 0.f;
#pragma unroll
    for (int c = 0; c < SS; ++c) qk += qr[c] * kr[c];
    qk *= 0.17677669529663687f;                       // 1/sqrt(S)
    float dist = 0.f;
    const float* qpr = qp_g + (size_t)n * 96 + h * 24;
    const float* kpr = kp_g + (size_t)nb * 24;
#pragma unroll
    for (int j = 0; j < 24; ++j) { float d = qpr[j] - kpr[j]; dist += d * d; }
    float b = 0.f;
    const float* pr = s_pair + k * PLD;
#pragma unroll 8
    for (int c = 0; c < CPC; ++c) b += pr[c] * s_Wb[c * HH + h];
    const float sp = log1pf(expf(gamma[h]));          // softplus
    const float scale = sp * (1.0f / 12.0f);          // * w_C/2 = (1/6)/2
    float lg = 0.57735026919f * (qk - scale * dist + b);  // w_L = sqrt(1/3)
    if (!s_mask[k]) lg = -1.0e9f;
    s_logits[k * HH + h] = lg;
  }
  __syncthreads();

  // softmax over k, per head (4 threads; K=32 serial is cheap)
  if (t < HH) {
    const int h = t;
    float m = -3.0e38f;
    for (int k = 0; k < KNB; ++k) m = fmaxf(m, s_logits[k * HH + h]);
    float e[KNB];
    float sum = 0.f;
    for (int k = 0; k < KNB; ++k) { e[k] = __expf(s_logits[k * HH + h] - m); sum += e[k]; }
    const float inv = 1.0f / sum;
    for (int k = 0; k < KNB; ++k)
      s_attn[k * HH + h] = s_mask[k] ? e[k] * inv : 0.f;
  }
  __syncthreads();

  __bf16* crow = cat + (size_t)n * NCAT;

  // pair_up: thread = pair column c (reuses LDS pair tile)
  {
    const int c = t;
    float a0 = 0.f, a1 = 0.f, a2 = 0.f, a3 = 0.f;
    for (int k = 0; k < KNB; ++k) {
      const float pv = s_pair[k * PLD + c];
      a0 += s_attn[k * HH + 0] * pv;
      a1 += s_attn[k * HH + 1] * pv;
      a2 += s_attn[k * HH + 2] * pv;
      a3 += s_attn[k * HH + 3] * pv;
    }
    crow[128 + 0 * CPC + c] = (__bf16)a0;
    crow[128 + 1 * CPC + c] = (__bf16)a1;
    crow[128 + 2 * CPC + c] = (__bf16)a2;
    crow[128 + 3 * CPC + c] = (__bf16)a3;
  }

  // local_up: thread -> (h, c); v lives in proj cols [160,192)
  {
    const int h = t >> 5, c = t & 31;
    float a = 0.f;
    for (int k = 0; k < KNB; ++k)
      a += s_attn[k * HH + h] * proj[(size_t)s_idx[k] * NPROJ + 160 + c];
    crow[h * SS + c] = (__bf16)a;
  }

  // point output: accumulate, then inverse frame (out_a = sum_b rot[b][a]*(x_b - tr_b))
  if (t < 96) {
    const int h = t / 24, j = t % 24;
    float a = 0.f;
    for (int k = 0; k < KNB; ++k)
      a += s_attn[k * HH + h] * vp_g[(size_t)s_idx[k] * 24 + j];
    s_pt[t] = a;
  }
  __syncthreads();
  if (t < 96) {
    const int h = t / 24, p = (t % 24) / 3, a = t % 3;
    const float x0 = s_pt[h * 24 + p * 3 + 0] - s_tr[0];
    const float x1 = s_pt[h * 24 + p * 3 + 1] - s_tr[1];
    const float x2 = s_pt[h * 24 + p * 3 + 2] - s_tr[2];
    crow[640 + t] = (__bf16)(s_rot[0 * 3 + a] * x0 + s_rot[1 * 3 + a] * x1 + s_rot[2 * 3 + a] * x2);
  }
}

// ---------------------------------------------------------------------------
extern "C" void kernel_launch(void* const* d_in, const int* in_sizes, int n_in,
                              void* d_out, int out_size, void* d_ws, size_t ws_size,
                              hipStream_t stream)
{
  const float* local  = (const float*)d_in[0];
  const float* pair   = (const float*)d_in[1];
  const float* frames = (const float*)d_in[2];
  const int*   nbr    = (const int*)d_in[3];
  const unsigned char* mask = (const unsigned char*)d_in[4];
  const float* Wq  = (const float*)d_in[5];
  const float* bq  = (const float*)d_in[6];
  const float* Wk  = (const float*)d_in[7];
  const float* bk  = (const float*)d_in[8];
  const float* Wv  = (const float*)d_in[9];
  const float* bv  = (const float*)d_in[10];
  const float* Wqp = (const float*)d_in[11];
  const float* bqp = (const float*)d_in[12];
  const float* Wkp = (const float*)d_in[13];
  const float* bkp = (const float*)d_in[14];
  const float* Wvp = (const float*)d_in[15];
  const float* bvp = (const float*)d_in[16];
  const float* Wb  = (const float*)d_in[17];
  const float* gamma = (const float*)d_in[18];
  const float* Wout  = (const float*)d_in[19];
  float* out = (float*)d_out;

  // Workspace carve-out (~40 MB total, 256B-aligned slices)
  char* w = (char*)d_ws;
  size_t off = 0;
  auto alloc = [&](size_t bytes) -> char* {
    char* p = w + off;
    off += (bytes + 255) & ~(size_t)255;
    return p;
  };
  __bf16* local_bf = (__bf16*)alloc((size_t)NNODE * CLC * 2);
  __bf16* WcatT    = (__bf16*)alloc((size_t)NPROJ * CLC * 2);
  float*  biascat  = (float*) alloc((size_t)NPROJ * 4);
  float*  proj     = (float*) alloc((size_t)NNODE * NPROJ * 4);
  float*  q_ln     = (float*) alloc((size_t)NNODE * 128 * 4);
  float*  k_ln     = (float*) alloc((size_t)NNODE * 32 * 4);
  float*  qp_g     = (float*) alloc((size_t)NNODE * 96 * 4);
  float*  kp_g     = (float*) alloc((size_t)NNODE * 24 * 4);
  float*  vp_g     = (float*) alloc((size_t)NNODE * 24 * 4);
  __bf16* cat      = (__bf16*)alloc((size_t)NNODE * NCAT * 2);
  __bf16* WoutT    = (__bf16*)alloc((size_t)CLC * NCAT * 2);
  (void)ws_size; (void)in_sizes; (void)n_in; (void)out_size;

  // 1) prep
  k_f32_to_bf16<<<(NNODE * CLC + 255) / 256, 256, 0, stream>>>(local, local_bf, NNODE * CLC);
  k_build_wcat<<<(NPROJ * CLC + 255) / 256, 256, 0, stream>>>(
      Wq, Wk, Wv, Wqp, Wkp, Wvp, bq, bk, bv, bqp, bkp, bvp, WcatT, biascat);
  k_build_woutT<<<(CLC * NCAT + 255) / 256, 256, 0, stream>>>(Wout, WoutT);

  // 2) proj = local @ Wcat + bias   (512 * 7 = 3584 strips of 16x48, 4 waves/block)
  wmma_gemm_bf16nt<<<(NNODE / 16) * (NPROJ / 48) / 4, 128, 0, stream>>>(
      local_bf, WcatT, biascat, proj, NNODE, NPROJ, CLC);

  // 3) LN + frames
  k_epilogue<<<NNODE / 256, 256, 0, stream>>>(proj, frames, q_ln, k_ln, qp_g, kp_g, vp_g);

  // 4) fused attention (single pass over pair)
  k_attention<<<NNODE, 128, 0, stream>>>(pair, frames, nbr, mask, proj,
                                         q_ln, k_ln, qp_g, kp_g, vp_g, Wb, gamma, cat);

  // 5) out = cat @ Wout   (512 * 8 = 4096 strips of 16x48)
  wmma_gemm_bf16nt<<<(NNODE / 16) * (CLC / 48) / 4, 128, 0, stream>>>(
      cat, WoutT, nullptr, out, NNODE, CLC, NCAT);
}
